// SuperpointMatching_56831007261098
// MI455X (gfx1250) — matrix operand order repaired and checked by reading, not compiled, not run on hardware
//
#include <hip/hip_runtime.h>
#include <hip/hip_bf16.h>

#define NS 4096
#define NT 4096
#define KD 256
#define NB 4
#define NCORR 256
#define CAP 4096
#define EPSV 1e-8f

typedef __attribute__((ext_vector_type(16))) _Float16 v16h;
typedef __attribute__((ext_vector_type(8)))  float    v8f;
typedef __attribute__((ext_vector_type(4)))  float    f32x4;

union HalfVec { v16h v; f32x4 f[2]; };

// ---------------------------------------------------------------------------
// Pass 1: L2-normalize rows (K=256) of src and tgt, emit f16. 1 wave = 1 row.
// ---------------------------------------------------------------------------
__global__ void norm_kernel(const float* __restrict__ src, const float* __restrict__ tgt,
                            _Float16* __restrict__ srcN, _Float16* __restrict__ tgtN) {
    const int wave = threadIdx.x >> 5;
    const int lane = threadIdx.x & 31;
    const int rid  = blockIdx.x * 8 + wave;             // 0 .. 2*NB*NS-1
    const float* inp;
    _Float16* outp;
    if (rid < NB * NS) { inp = src + (size_t)rid * KD;               outp = srcN + (size_t)rid * KD; }
    else               { int r2 = rid - NB * NS;
                         inp = tgt + (size_t)r2 * KD;                outp = tgtN + (size_t)r2 * KD; }
    f32x4 x0 = *(const f32x4*)(inp + lane * 8);
    f32x4 x1 = *(const f32x4*)(inp + lane * 8 + 4);
    float ss = x0.x*x0.x + x0.y*x0.y + x0.z*x0.z + x0.w*x0.w
             + x1.x*x1.x + x1.y*x1.y + x1.z*x1.z + x1.w*x1.w;
#pragma unroll
    for (int m = 1; m < 32; m <<= 1) ss += __shfl_xor(ss, m, 32);
    const float inv = 1.0f / fmaxf(sqrtf(ss), 1e-12f);
    union { _Float16 h[8]; f32x4 f; } o;
    o.h[0] = (_Float16)(x0.x * inv); o.h[1] = (_Float16)(x0.y * inv);
    o.h[2] = (_Float16)(x0.z * inv); o.h[3] = (_Float16)(x0.w * inv);
    o.h[4] = (_Float16)(x1.x * inv); o.h[5] = (_Float16)(x1.y * inv);
    o.h[6] = (_Float16)(x1.z * inv); o.h[7] = (_Float16)(x1.w * inv);
    *(f32x4*)(outp + lane * 8) = o.f;
}

// ---------------------------------------------------------------------------
// WMMA core: one wave computes a 32(M) x 64(N) strip with K=256.
//   A fragment (16x32 f16): lane = row M=l16; lanes<16 hold K {0..7,16..23},
//   lanes>=16 hold K {8..15,24..31} -> two contiguous 16B loads.
//   B fragment (32x16 f16): lane = col N=l16 (T-row t0+l16); lanes<16 hold
//   K {0..7,16..23}, lanes>=16 hold K {8..15,24..31} -> two 16B loads.
// Per K-step: 12 b128 loads feed 8 v_wmma (1.5 loads/WMMA); all loads of a
// step are issued before its WMMAs so the batch hides load latency.
// ---------------------------------------------------------------------------
__device__ __forceinline__ void wave_gemm_32x64(const _Float16* __restrict__ S,
                                                const _Float16* __restrict__ T,
                                                int s0, int t0, v8f acc[2][4]) {
    const int lane = threadIdx.x & 31;
    const int half = lane >> 4;
    const int l16  = lane & 15;
    const _Float16* arow0 = S + (size_t)(s0 + l16) * KD;
    const _Float16* arow1 = S + (size_t)(s0 + 16 + l16) * KD;
    const int ac0 = half ? 8  : 0;
    const int ac1 = half ? 24 : 16;
    const int bc  = half ? 8 : 0;
    const _Float16* brow0 = T + (size_t)(t0 + l16) * KD + bc;
#pragma unroll
    for (int kk = 0; kk < KD; kk += 32) {
        HalfVec a0, a1, bm[4];
        a0.f[0] = *(const f32x4*)(arow0 + kk + ac0);
        a0.f[1] = *(const f32x4*)(arow0 + kk + ac1);
        a1.f[0] = *(const f32x4*)(arow1 + kk + ac0);
        a1.f[1] = *(const f32x4*)(arow1 + kk + ac1);
#pragma unroll
        for (int j = 0; j < 4; ++j) {
            const _Float16* bj = brow0 + (size_t)j * 16 * KD + kk;
            bm[j].f[0] = *(const f32x4*)(bj);
            bm[j].f[1] = *(const f32x4*)(bj + 16);
        }
#pragma unroll
        for (int j = 0; j < 4; ++j)
            acc[0][j] = __builtin_amdgcn_wmma_f32_16x16x32_f16(false, a0.v, false, bm[j].v,
                                                               (short)0, acc[0][j], false, false);
#pragma unroll
        for (int j = 0; j < 4; ++j)
            acc[1][j] = __builtin_amdgcn_wmma_f32_16x16x32_f16(false, a1.v, false, bm[j].v,
                                                               (short)0, acc[1][j], false, false);
    }
}

// C/D layout: element r of v8f, lane L -> M = r + 8*(L>=16), N = L%16.

// ---------------------------------------------------------------------------
// Pass 2: GEMM + exp, emit deterministic per-block partial row/col sums.
// Block = 256 threads = 8 waves covering 256(M) x 64(N).
// ---------------------------------------------------------------------------
__global__ void gemm_sums_kernel(const _Float16* __restrict__ srcN, const _Float16* __restrict__ tgtN,
                                 float* __restrict__ rowpart, float* __restrict__ colpart) {
    __shared__ float scol[8][64];
    const int b    = blockIdx.z;
    const int wave = threadIdx.x >> 5;
    const int lane = threadIdx.x & 31;
    const int half = lane >> 4;
    const int l16  = lane & 15;
    const int s0   = blockIdx.y * 256 + wave * 32;
    const int t0   = blockIdx.x * 64;
    const _Float16* S = srcN + ((size_t)b << 20);
    const _Float16* T = tgtN + ((size_t)b << 20);
    const v8f vzero = {0.f,0.f,0.f,0.f,0.f,0.f,0.f,0.f};
    v8f acc[2][4];
#pragma unroll
    for (int i = 0; i < 2; ++i)
#pragma unroll
        for (int j = 0; j < 4; ++j) acc[i][j] = vzero;
    wave_gemm_32x64(S, T, s0, t0, acc);
    float corr[2][4][8];
#pragma unroll
    for (int i = 0; i < 2; ++i)
#pragma unroll
        for (int j = 0; j < 4; ++j)
#pragma unroll
            for (int r = 0; r < 8; ++r)
                corr[i][j][r] = __expf(2.0f * acc[i][j][r] - 2.0f);

    // row partial sums over this block's 64 columns (reduce across 16 lanes)
#pragma unroll
    for (int i = 0; i < 2; ++i)
#pragma unroll
        for (int r = 0; r < 8; ++r) {
            float v = corr[i][0][r] + corr[i][1][r] + corr[i][2][r] + corr[i][3][r];
            v += __shfl_xor(v, 1, 32);
            v += __shfl_xor(v, 2, 32);
            v += __shfl_xor(v, 4, 32);
            v += __shfl_xor(v, 8, 32);
            if (l16 == 0)
                rowpart[(((size_t)b * gridDim.x + blockIdx.x) << 12) + s0 + i * 16 + r + 8 * half] = v;
        }
    // col partial sums over this block's 256 rows
#pragma unroll
    for (int j = 0; j < 4; ++j) {
        float v = 0.f;
#pragma unroll
        for (int i = 0; i < 2; ++i)
#pragma unroll
            for (int r = 0; r < 8; ++r) v += corr[i][j][r];
        v += __shfl_xor(v, 16, 32);
        if (half == 0) scol[wave][j * 16 + l16] = v;
    }
    __syncthreads();
    if (threadIdx.x < 64) {
        float v = 0.f;
#pragma unroll
        for (int w = 0; w < 8; ++w) v += scol[w][threadIdx.x];
        colpart[(((size_t)b * gridDim.y + blockIdx.y) << 12) + t0 + threadIdx.x] = v;
    }
}

// ---------------------------------------------------------------------------
// Pass 3: reduce partials -> 1/(sum+eps) for rows and cols.
// ---------------------------------------------------------------------------
__global__ void reduce_inv_kernel(const float* __restrict__ rowpart, const float* __restrict__ colpart,
                                  float* __restrict__ rowinv, float* __restrict__ colinv) {
    const int gid = blockIdx.x * blockDim.x + threadIdx.x;     // 0 .. 2*NB*NS-1
    if (gid < NB * NS) {
        const int b = gid >> 12, s = gid & 4095;
        float acc = 0.f;
        for (int g = 0; g < 64; ++g) acc += rowpart[(((size_t)b * 64 + g) << 12) + s];
        rowinv[gid] = 1.0f / (acc + EPSV);
    } else {
        const int g2 = gid - NB * NS;
        const int b = g2 >> 12, t = g2 & 4095;
        float acc = 0.f;
        for (int g = 0; g < 16; ++g) acc += colpart[(((size_t)b * 16 + g) << 12) + t];
        colinv[g2] = 1.0f / (acc + EPSV);
    }
}

// Dual-normalized scores for one wave's 32x64 strip (recompute GEMM).
__device__ __forceinline__ void wave_vals(const _Float16* __restrict__ S, const _Float16* __restrict__ T,
                                          const float* __restrict__ rowinv, const float* __restrict__ colinv,
                                          int b, int s0, int t0, float val[2][4][8]) {
    const int lane = threadIdx.x & 31;
    const int half = lane >> 4;
    const int l16  = lane & 15;
    const v8f vzero = {0.f,0.f,0.f,0.f,0.f,0.f,0.f,0.f};
    v8f acc[2][4];
#pragma unroll
    for (int i = 0; i < 2; ++i)
#pragma unroll
        for (int j = 0; j < 4; ++j) acc[i][j] = vzero;
    wave_gemm_32x64(S, T, s0, t0, acc);
    float rinv[2][8];
#pragma unroll
    for (int i = 0; i < 2; ++i)
#pragma unroll
        for (int r = 0; r < 8; ++r)
            rinv[i][r] = rowinv[(b << 12) + s0 + i * 16 + r + 8 * half];
#pragma unroll
    for (int j = 0; j < 4; ++j) {
        const float cinv = colinv[(b << 12) + t0 + j * 16 + l16];
#pragma unroll
        for (int i = 0; i < 2; ++i)
#pragma unroll
            for (int r = 0; r < 8; ++r) {
                const float c = __expf(2.0f * acc[i][j][r] - 2.0f);
                val[i][j][r] = c * c * rinv[i][r] * cinv;  // corr^2/((rowsum+eps)(colsum+eps))
            }
    }
}

// ---------------------------------------------------------------------------
// Pass 4: histogram of float-bit buckets (monotone for positive floats).
// ---------------------------------------------------------------------------
__global__ void gemm_hist_kernel(const _Float16* __restrict__ srcN, const _Float16* __restrict__ tgtN,
                                 const float* __restrict__ rowinv, const float* __restrict__ colinv,
                                 unsigned* __restrict__ hist) {
    __shared__ unsigned lhist[4096];
    for (int i = threadIdx.x; i < 4096; i += 256) lhist[i] = 0u;
    __syncthreads();
    const int b    = blockIdx.z;
    const int wave = threadIdx.x >> 5;
    const int s0   = blockIdx.y * 256 + wave * 32;
    const int t0   = blockIdx.x * 64;
    const _Float16* S = srcN + ((size_t)b << 20);
    const _Float16* T = tgtN + ((size_t)b << 20);
    float val[2][4][8];
    wave_vals(S, T, rowinv, colinv, b, s0, t0, val);
#pragma unroll
    for (int i = 0; i < 2; ++i)
#pragma unroll
        for (int j = 0; j < 4; ++j)
#pragma unroll
            for (int r = 0; r < 8; ++r) {
                unsigned bin = __float_as_uint(val[i][j][r]) >> 19;
                if (bin > 4095u) bin = 4095u;
                atomicAdd(&lhist[bin], 1u);
            }
    __syncthreads();
    for (int i = threadIdx.x; i < 4096; i += 256)
        if (lhist[i]) atomicAdd(&hist[(b << 12) + i], lhist[i]);
}

// Pass 5: per-batch suffix scan -> bucket threshold covering >= 256 elements.
__global__ void scan_kernel(const unsigned* __restrict__ hist, float* __restrict__ thresh) {
    const int b = threadIdx.x;
    if (b >= NB) return;
    unsigned cum = 0;
    float th = 0.0f;
    for (int bin = 4095; bin >= 0; --bin) {
        cum += hist[(b << 12) + bin];
        if (cum >= NCORR) { th = __uint_as_float((unsigned)bin << 19); break; }
    }
    thresh[b] = th;
}

// ---------------------------------------------------------------------------
// Pass 6: recompute + compact candidates >= threshold.
// ---------------------------------------------------------------------------
__global__ void gemm_compact_kernel(const _Float16* __restrict__ srcN, const _Float16* __restrict__ tgtN,
                                    const float* __restrict__ rowinv, const float* __restrict__ colinv,
                                    const float* __restrict__ thresh,
                                    unsigned* __restrict__ cnt,
                                    float* __restrict__ candv, unsigned* __restrict__ candi) {
    const int b    = blockIdx.z;
    const int wave = threadIdx.x >> 5;
    const int lane = threadIdx.x & 31;
    const int half = lane >> 4;
    const int l16  = lane & 15;
    const int s0   = blockIdx.y * 256 + wave * 32;
    const int t0   = blockIdx.x * 64;
    const _Float16* S = srcN + ((size_t)b << 20);
    const _Float16* T = tgtN + ((size_t)b << 20);
    float val[2][4][8];
    wave_vals(S, T, rowinv, colinv, b, s0, t0, val);
    const float th = thresh[b];
#pragma unroll
    for (int i = 0; i < 2; ++i)
#pragma unroll
        for (int j = 0; j < 4; ++j)
#pragma unroll
            for (int r = 0; r < 8; ++r) {
                if (val[i][j][r] >= th) {
                    const unsigned p = atomicAdd(&cnt[b], 1u);
                    if (p < (unsigned)CAP) {
                        candv[b * CAP + p] = val[i][j][r];
                        candi[b * CAP + p] =
                            ((unsigned)(s0 + i * 16 + r + 8 * half) << 12) |
                            (unsigned)(t0 + j * 16 + l16);
                    }
                }
            }
}

// ---------------------------------------------------------------------------
// Pass 7: per-batch sorted top-256 from <=4096 candidates (iterative argmax).
// ---------------------------------------------------------------------------
__global__ void select_kernel(const unsigned* __restrict__ cnt,
                              const float* __restrict__ candv, const unsigned* __restrict__ candi,
                              float* __restrict__ out) {
    __shared__ float    sv[CAP];
    __shared__ unsigned si[CAP];
    __shared__ float    rv[1024];
    __shared__ int      ri[1024];
    const int b   = blockIdx.x;
    const int tid = threadIdx.x;
    const unsigned cb = cnt[b];
    const int n = (cb < (unsigned)CAP) ? (int)cb : CAP;
    for (int i = tid; i < CAP; i += 1024) {
        sv[i] = (i < n) ? candv[b * CAP + i] : -1.0f;
        si[i] = (i < n) ? candi[b * CAP + i] : 0u;
    }
    __syncthreads();
    for (int it = 0; it < NCORR; ++it) {
        float bm = -2.0f; int bp = 0;
        for (int i = tid; i < CAP; i += 1024)
            if (sv[i] > bm) { bm = sv[i]; bp = i; }
        rv[tid] = bm; ri[tid] = bp;
        __syncthreads();
        for (int off = 512; off > 0; off >>= 1) {
            if (tid < off && rv[tid + off] > rv[tid]) { rv[tid] = rv[tid + off]; ri[tid] = ri[tid + off]; }
            __syncthreads();
        }
        if (tid == 0) {
            const int pos = ri[0];
            const unsigned fi = si[pos];
            out[b * NCORR + it]                    = (float)(fi >> 12);    // src index
            out[NB * NCORR + b * NCORR + it]       = (float)(fi & 4095u);  // tgt index
            out[2 * NB * NCORR + b * NCORR + it]   = rv[0];                // score
            sv[pos] = -3.0f;
        }
        __syncthreads();
    }
}

extern "C" void kernel_launch(void* const* d_in, const int* in_sizes, int n_in,
                              void* d_out, int out_size, void* d_ws, size_t ws_size,
                              hipStream_t stream) {
    (void)in_sizes; (void)n_in; (void)out_size; (void)ws_size;
    const float* src = (const float*)d_in[0];
    const float* tgt = (const float*)d_in[1];
    float* out = (float*)d_out;
    char* ws = (char*)d_ws;
    size_t off = 0;
    auto alloc = [&](size_t bytes) -> void* {
        void* p = ws + off;
        off += (bytes + 255) & ~(size_t)255;
        return p;
    };
    _Float16* srcN  = (_Float16*)alloc((size_t)NB * NS * KD * 2);     // 8.4 MB
    _Float16* tgtN  = (_Float16*)alloc((size_t)NB * NT * KD * 2);     // 8.4 MB
    float* rowpart  = (float*)alloc((size_t)NB * 64 * NS * 4);        // 4 MB
    float* colpart  = (float*)alloc((size_t)NB * 16 * NT * 4);        // 1 MB
    float* rowinv   = (float*)alloc((size_t)NB * NS * 4);
    float* colinv   = (float*)alloc((size_t)NB * NT * 4);
    unsigned* hist  = (unsigned*)alloc((size_t)NB * 4096 * 4);
    float* thresh   = (float*)alloc(NB * 4);
    unsigned* cnt   = (unsigned*)alloc(NB * 4);
    float* candv    = (float*)alloc((size_t)NB * CAP * 4);
    unsigned* candi = (unsigned*)alloc((size_t)NB * CAP * 4);

    hipMemsetAsync(hist, 0, (size_t)NB * 4096 * 4, stream);
    hipMemsetAsync(cnt,  0, (size_t)NB * 4, stream);

    norm_kernel<<<(NB * (NS + NT)) / 8, 256, 0, stream>>>(src, tgt, srcN, tgtN);
    dim3 g(NT / 64, NS / 256, NB);
    gemm_sums_kernel<<<g, 256, 0, stream>>>(srcN, tgtN, rowpart, colpart);
    reduce_inv_kernel<<<(NB * (NS + NT)) / 256, 256, 0, stream>>>(rowpart, colpart, rowinv, colinv);
    gemm_hist_kernel<<<g, 256, 0, stream>>>(srcN, tgtN, rowinv, colinv, hist);
    scan_kernel<<<1, 32, 0, stream>>>(hist, thresh);
    gemm_compact_kernel<<<g, 256, 0, stream>>>(srcN, tgtN, rowinv, colinv, thresh, cnt, candv, candi);
    select_kernel<<<NB, 1024, 0, stream>>>(cnt, candv, candi, out);
}
